// Match_HGT_58239756533859
// MI455X (gfx1250) — matrix operand order repaired and checked by reading, not compiled, not run on hardware
//
#include <hip/hip_runtime.h>
#include <hip/hip_bf16.h>
#include <math.h>

// ---------------------------------------------------------------------------
// HGT forward for MI455X (gfx1250, wave32, WMMA).
//   - All dense linears: bf16 WMMA (v_wmma_f32_16x16x32_bf16), f32 accumulate.
//   - GEMM tiles staged with GLOBAL_LOAD_ASYNC_TO_LDS_B128 + s_wait_asynccnt
//     (double-buffered LDS). Builtin signature confirmed by compiler probe:
//     (v4i32 AS1* gsrc, v4i32 AS3* ldst, imm offset, imm cpol).
//   - Relation matrices folded into K/V weights (per-node, not per-edge).
//   - Edge phase: wave-per-edge gather + shfl_xor head reduction +
//     ordered-u32 atomicMax segment-max + f32 atomic-add segment-sum/scatter.
// ---------------------------------------------------------------------------

#define DEV __device__ __forceinline__

typedef __attribute__((ext_vector_type(16))) __bf16 v16bf;
typedef __attribute__((ext_vector_type(8)))  float  v8f;

struct U128x2 { uint4 a, b; };
struct __attribute__((aligned(8))) U16x4 { unsigned short x, y, z, w; };

static const int NL  = 100000, NA = 20000;
static const int FL  = 768,    FA = 300, FAp = 320;   // FA padded to mult of 32
static const int C   = 128;
static const int ECC = 600000, ECA = 300000, EAC = 300000;

// -------- CDNA5 async global->LDS copy (ASYNCcnt path) ---------------------
#if defined(__has_builtin)
#  if __has_builtin(__builtin_amdgcn_global_load_async_to_lds_b128)
#    define HAVE_ASYNC_LDS 1
#  endif
#endif
#ifndef HAVE_ASYNC_LDS
#  define HAVE_ASYNC_LDS 0
#endif

#if HAVE_ASYNC_LDS
typedef __attribute__((vector_size(16))) int v4i_t;
typedef __attribute__((address_space(1))) v4i_t gas_v4i;   // global (prints as __device__)
typedef __attribute__((address_space(3))) v4i_t las_v4i;   // LDS
DEV void async_copy16(const unsigned short* g, unsigned short* l) {
  // 16 bytes per lane, wave-level async transfer into LDS (ASYNCcnt)
  __builtin_amdgcn_global_load_async_to_lds_b128((gas_v4i*)g, (las_v4i*)l, 0, 0);
}
#endif

DEV void wait_async0() {
#if defined(__has_builtin)
#  if __has_builtin(__builtin_amdgcn_s_wait_asynccnt)
  __builtin_amdgcn_s_wait_asynccnt(0);
  return;
#  endif
#endif
#if HAVE_ASYNC_LDS
  asm volatile("s_wait_asynccnt 0" ::: "memory");
#endif
}

DEV v16bf frag16(uint4 lo, uint4 hi) {
  U128x2 t; t.a = lo; t.b = hi;
  return __builtin_bit_cast(v16bf, t);
}

DEV unsigned short f2bf(float f) {                 // round-to-nearest-even
  unsigned u = __float_as_uint(f);
  unsigned r = u + 0x7FFFu + ((u >> 16) & 1u);
  return (unsigned short)(r >> 16);
}
DEV float bf2f(unsigned short h) { return __uint_as_float(((unsigned)h) << 16); }

// order-preserving float <-> u32 encoding for atomic segment-max
DEV unsigned encOrd(float x) {
  unsigned u = __float_as_uint(x);
  return (u & 0x80000000u) ? ~u : (u | 0x80000000u);
}
DEV float decOrd(unsigned e) {
  return (e & 0x80000000u) ? __uint_as_float(e & 0x7FFFFFFFu)
                           : __uint_as_float(~e);
}

// ---------------------------------------------------------------------------
// Generic f32->bf16 convert with zero padding (pads K dims to mult of 32).
// ---------------------------------------------------------------------------
__global__ void __launch_bounds__(256)
cvt_pad_bf16(const float* __restrict__ src, unsigned short* __restrict__ dst,
             int srcRows, int srcCols, int dstRows, int dstCols)
{
  long idx = blockIdx.x * 256L + threadIdx.x;
  long n = (long)dstRows * dstCols;
  if (idx >= n) return;
  int r = (int)(idx / dstCols), c = (int)(idx % dstCols);
  float v = (r < srcRows && c < srcCols) ? src[(long)r * srcCols + c] : 0.f;
  dst[idx] = f2bf(v);
}

// ---------------------------------------------------------------------------
// Fold per-head relation matrix into a 128x128 weight (and its bias):
//   Wout[:, h*16+f] = sum_d W[:, h*16+d] * rel[h,d,f]     (bf16 out)
//   bout[h*16+f]    = sum_d b[h*16+d]    * rel[h,d,f]     (f32 out, row 128)
// ---------------------------------------------------------------------------
__global__ void __launch_bounds__(256)
fold_rel(const float* __restrict__ W, const float* __restrict__ b,
         const float* __restrict__ rel,
         unsigned short* __restrict__ Wout, float* __restrict__ bout)
{
  int idx = blockIdx.x * 256 + threadIdx.x;
  if (idx >= 129 * 128) return;
  int row = idx >> 7, col = idx & 127;
  int h = col >> 4, f = col & 15;
  const float* rp = rel + h * 256 + f;     // rel[h][d][f], stride 16 over d
  if (row < 128) {
    const float* wp = W + row * 128 + h * 16;
    float s = 0.f;
    #pragma unroll
    for (int d2 = 0; d2 < 16; ++d2) s += wp[d2] * rp[d2 * 16];
    Wout[row * 128 + col] = f2bf(s);
  } else {
    const float* bp = b + h * 16;
    float s = 0.f;
    #pragma unroll
    for (int d2 = 0; d2 < 16; ++d2) s += bp[d2] * rp[d2 * 16];
    bout[col] = s;
  }
}

// ---------------------------------------------------------------------------
// bf16 WMMA GEMM:  out[M,128] = act(A[M,K] @ W[K,128] + bias)
// K multiple of 32, lda == K. Block = 256 threads = 8 waves; each wave owns a
// 16x128 strip (8 wmma accumulators). Fragment packing per CDNA5 ISA 7.12.2.
// Async path: double-buffered LDS, tile i+1 copied by the async unit while
// tile i feeds WMMAs; one barrier per K-chunk.
// ---------------------------------------------------------------------------
__global__ void __launch_bounds__(256)
gemm_bf16_wmma(const unsigned short* __restrict__ A, int M, int K,
               const unsigned short* __restrict__ W, const float* __restrict__ bias,
               float* __restrict__ outF, unsigned short* __restrict__ outB, int relu)
{
#if HAVE_ASYNC_LDS
  __shared__ unsigned short ldsA[2][128 * 32];   // 2 x 8 KB
  __shared__ unsigned short ldsB[2][32 * 128];   // 2 x 8 KB
#else
  __shared__ unsigned short ldsA[1][128 * 32];
  __shared__ unsigned short ldsB[1][32 * 128];
#endif

  const int tid  = threadIdx.x;
  const int wave = tid >> 5;
  const int lane = tid & 31;
  const long row0 = (long)blockIdx.x * 128;

  v8f acc[8];
  {
    v8f z;
    #pragma unroll
    for (int j = 0; j < 8; ++j) z[j] = 0.f;
    #pragma unroll
    for (int i = 0; i < 8; ++i) acc[i] = z;
  }

  const int ar = tid >> 1;            // A tile row loaded by this thread
  const int ah = (tid & 1) * 16;      // 16-element half of the 32-wide row
  const int bk = tid >> 3;            // B tile row (K)
  const int bs = (tid & 7) * 16;      // B column segment
  const long arow = row0 + ar;

  // A fragment addressing (ISA 7.12.2): lane l -> row l&15,
  // halves {0..7} = K[8*(l>>4)+0..7], halves {8..15} = K[16+8*(l>>4)+0..7].
  // B fragment: lane l holds N-col l&15, K in the same order as A.
  auto compute = [&](const unsigned short* tA, const unsigned short* tB) {
    const int m   = lane & 15;
    const int hig = (lane >> 4) * 8;
    uint4 fa0 = *(const uint4*)&tA[(wave * 16 + m) * 32 + hig];
    uint4 fa1 = *(const uint4*)&tA[(wave * 16 + m) * 32 + 16 + hig];
    v16bf afrag = frag16(fa0, fa1);
    #pragma unroll
    for (int nt = 0; nt < 8; ++nt) {
      v16bf bfrag;
      #pragma unroll
      for (int s = 0; s < 16; ++s) {
        int kk = (s < 8) ? hig + s : 16 + hig + (s - 8);
        bfrag[s] = __builtin_bit_cast(__bf16, tB[kk * 128 + nt * 16 + m]);
      }
      acc[nt] = __builtin_amdgcn_wmma_f32_16x16x32_bf16(
          false, afrag, false, bfrag, (short)0, acc[nt], false, false);
    }
  };

#if HAVE_ASYNC_LDS
  auto stage = [&](int buf, int k0) {
    if (arow < M) {      // OOB rows: stale LDS only feeds masked-out outputs
      const unsigned short* g = A + arow * (long)K + k0 + ah;
      async_copy16(g,     &ldsA[buf][ar * 32 + ah]);
      async_copy16(g + 8, &ldsA[buf][ar * 32 + ah + 8]);
    }
    const unsigned short* w = W + (long)(k0 + bk) * 128 + bs;
    async_copy16(w,     &ldsB[buf][bk * 128 + bs]);
    async_copy16(w + 8, &ldsB[buf][bk * 128 + bs + 8]);
  };
  const int nch = K >> 5;
  stage(0, 0);
  for (int i = 0; i < nch; ++i) {
    wait_async0();        // tile i resident (tile i+1 copy not yet issued)
    __syncthreads();      // publish tile i; proves prev buffer fully consumed
    if (i + 1 < nch) stage((i + 1) & 1, (i + 1) * 32);
    compute(ldsA[i & 1], ldsB[i & 1]);
  }
#else
  for (int k0 = 0; k0 < K; k0 += 32) {
    uint4 alo = {0, 0, 0, 0}, ahi = {0, 0, 0, 0};
    if (arow < M) {
      const uint4* gp = (const uint4*)(A + arow * (long)K + k0 + ah);
      alo = gp[0]; ahi = gp[1];
      if (k0 + 32 < K)                                   // global_prefetch_b8
        __builtin_prefetch((const void*)(A + arow * (long)K + k0 + 32 + ah), 0, 1);
    }
    const uint4* wp = (const uint4*)(W + (long)(k0 + bk) * 128 + bs);
    uint4 blo = wp[0], bhi = wp[1];

    __syncthreads();
    *(uint4*)&ldsA[0][ar * 32 + ah]      = alo;
    *(uint4*)&ldsA[0][ar * 32 + ah + 8]  = ahi;
    *(uint4*)&ldsB[0][bk * 128 + bs]     = blo;
    *(uint4*)&ldsB[0][bk * 128 + bs + 8] = bhi;
    __syncthreads();
    compute(ldsA[0], ldsB[0]);
  }
#endif

  // Epilogue. C/D layout: lane l, vgpr r -> M = r + 8*(l>>4), N = l&15.
  const int n16 = lane & 15;
  const int hig = lane >> 4;
  #pragma unroll
  for (int nt = 0; nt < 8; ++nt) {
    int n = nt * 16 + n16;
    float bv = bias ? bias[n] : 0.f;
    #pragma unroll
    for (int r = 0; r < 8; ++r) {
      long row = row0 + wave * 16 + hig * 8 + r;
      if (row < M) {
        float v = acc[nt][r] + bv;
        if (relu) v = v > 0.f ? v : 0.f;
        if (outF) outF[row * 128 + n] = v;
        if (outB) outB[row * 128 + n] = f2bf(v);
      }
    }
  }
}

// ---------------------------------------------------------------------------
// Edge phase. Wave-per-edge; lane l owns channels [4l, 4l+4) (head = l>>2).
// ---------------------------------------------------------------------------
__global__ void __launch_bounds__(256)
edge_logits_max(const int* __restrict__ ei, int E,
                const unsigned short* __restrict__ krel,
                const unsigned short* __restrict__ q,
                const float* __restrict__ prel,
                float* __restrict__ logits, unsigned int* __restrict__ amax)
{
  long gid = blockIdx.x * 256L + threadIdx.x;
  long e = gid >> 5;
  int lane = (int)(gid & 31);
  if (e >= E) return;
  int s = ei[e], d = ei[E + e];
  U16x4 kv = *(const U16x4*)(krel + (long)s * 128 + lane * 4);
  U16x4 qv = *(const U16x4*)(q    + (long)d * 128 + lane * 4);
  float p = bf2f(kv.x) * bf2f(qv.x) + bf2f(kv.y) * bf2f(qv.y)
          + bf2f(kv.z) * bf2f(qv.z) + bf2f(kv.w) * bf2f(qv.w);
  p += __shfl_xor(p, 1, 32);                 // reduce 4 lanes of one head
  p += __shfl_xor(p, 2, 32);
  if ((lane & 3) == 0) {
    int h = lane >> 2;
    float lg = p * prel[h] * 0.25f;          // 1/sqrt(D), D=16
    logits[e * 8 + h] = lg;
    atomicMax(&amax[(long)d * 8 + h], encOrd(lg));
  }
}

__global__ void __launch_bounds__(256)
edge_exp(const int* __restrict__ ei, int E, float* __restrict__ lg,
         const unsigned int* __restrict__ amax, float* __restrict__ den)
{
  long idx = blockIdx.x * 256L + threadIdx.x;
  if (idx >= (long)E * 8) return;
  long e = idx >> 3;
  int h = (int)(idx & 7);
  int d = ei[E + e];
  float mx = decOrd(amax[(long)d * 8 + h]);
  float ex = __expf(lg[idx] - mx);
  lg[idx] = ex;                               // overwrite logits with exp
  __hip_atomic_fetch_add(&den[(long)d * 8 + h], ex,
                         __ATOMIC_RELAXED, __HIP_MEMORY_SCOPE_AGENT);
}

__global__ void __launch_bounds__(256)
edge_scatter(const int* __restrict__ ei, int E, const float* __restrict__ ex,
             const float* __restrict__ den, const unsigned short* __restrict__ vrel,
             float* __restrict__ acc)
{
  long gid = blockIdx.x * 256L + threadIdx.x;
  long e = gid >> 5;
  int lane = (int)(gid & 31);
  if (e >= E) return;
  int s = ei[e], d = ei[E + e];
  int h = lane >> 2;
  float alpha = ex[e * 8 + h] / den[(long)d * 8 + h];
  U16x4 vv = *(const U16x4*)(vrel + (long)s * 128 + lane * 4);
  float* ap = acc + (long)d * 128 + lane * 4;
  __hip_atomic_fetch_add(&ap[0], alpha * bf2f(vv.x), __ATOMIC_RELAXED, __HIP_MEMORY_SCOPE_AGENT);
  __hip_atomic_fetch_add(&ap[1], alpha * bf2f(vv.y), __ATOMIC_RELAXED, __HIP_MEMORY_SCOPE_AGENT);
  __hip_atomic_fetch_add(&ap[2], alpha * bf2f(vv.z), __ATOMIC_RELAXED, __HIP_MEMORY_SCOPE_AGENT);
  __hip_atomic_fetch_add(&ap[3], alpha * bf2f(vv.w), __ATOMIC_RELAXED, __HIP_MEMORY_SCOPE_AGENT);
}

// mean over relations + exact GELU, emit bf16 for the a-linear GEMM
__global__ void __launch_bounds__(256)
combine_gelu(const float* __restrict__ a0, const float* __restrict__ a1,
             float scale, unsigned short* __restrict__ out, long n)
{
  long i = blockIdx.x * 256L + threadIdx.x;
  if (i >= n) return;
  float x = a0[i];
  if (a1) x += a1[i];
  x *= scale;
  float g = 0.5f * x * (1.f + erff(x * 0.70710678118654752f));
  out[i] = f2bf(g);
}

// new_x = sigmoid(gate)*o + (1-sigmoid(gate))*x ; refresh f32 state + bf16 mirror
__global__ void __launch_bounds__(256)
skip_gate(const float* __restrict__ o, const float* __restrict__ gate,
          float* __restrict__ sF, unsigned short* __restrict__ sB, long n)
{
  long i = blockIdx.x * 256L + threadIdx.x;
  if (i >= n) return;
  float g = 1.f / (1.f + __expf(-gate[0]));
  float v = g * o[i] + (1.f - g) * sF[i];
  sF[i] = v;
  sB[i] = f2bf(v);
}

__global__ void __launch_bounds__(256)
gather_docs(const float* __restrict__ sL, const int* __restrict__ A,
            const int* __restrict__ B, float* __restrict__ out)
{
  int i = blockIdx.x * 256 + threadIdx.x;
  if (i >= 2 * 256 * 128) return;
  int half = i >> 15, r = (i >> 7) & 255, c = i & 127;
  int node = half ? B[r] : A[r];
  out[i] = sL[(long)node * 128 + c];
}

// ---------------------------------------------------------------------------
// Host orchestration.
// Param leaves (d_in[7..]) in jax pytree order (dicts sorted by key):
// conv[0], conv[1] (27 leaves each: a.art.{W,b}, a.law.{W,b}, k.*, q.*,
// rel.cites.{a,m,p}, rel.referred.{a,m,p}, rel.refers.{a,m,p}, skip.art,
// skip.law, v.art.{W,b}, v.law.{W,b}), then post (art W1,b1,W2,b2, law ...),
// then pre (art ..., law ...).
// ---------------------------------------------------------------------------
extern "C" void kernel_launch(void* const* d_in, const int* in_sizes, int n_in,
                              void* d_out, int out_size, void* d_ws, size_t ws_size,
                              hipStream_t stream)
{
  (void)in_sizes; (void)n_in; (void)out_size; (void)ws_size;

  const float* x_law       = (const float*)d_in[0];
  const float* x_art       = (const float*)d_in[1];
  const int*   ei_cites    = (const int*)d_in[2];
  const int*   ei_refers   = (const int*)d_in[3];
  const int*   ei_referred = (const int*)d_in[4];
  const int*   docA        = (const int*)d_in[5];
  const int*   docB        = (const int*)d_in[6];
  auto P = [&](int i) { return (const float*)d_in[7 + i]; };

  // within-layer leaf offsets
  const int A_ART_W=0, A_ART_B=1, A_LAW_W=2, A_LAW_B=3;
  const int K_ART_W=4, K_ART_B=5, K_LAW_W=6, K_LAW_B=7;
  const int Q_ART_W=8, Q_ART_B=9, Q_LAW_W=10, Q_LAW_B=11;
  const int R_CIT_A=12, R_CIT_M=13, R_CIT_P=14;
  const int R_RED_A=15, R_RED_M=16, R_RED_P=17;   // "referred"
  const int R_REF_A=18, R_REF_M=19, R_REF_P=20;   // "refers"
  const int SKIP_ART=21, SKIP_LAW=22;
  const int V_ART_W=23, V_ART_B=24, V_LAW_W=25, V_LAW_B=26;
  const int POST = 54, PRE = 62;

  // --- workspace bump allocator --------------------------------------------
  char* wp = (char*)d_ws;
  auto alloc = [&](size_t bytes) -> void* {
    void* r = (void*)wp;
    wp += (bytes + 255) & ~(size_t)255;
    return r;
  };
  typedef unsigned short u16;

  u16* xlawB = (u16*)alloc((size_t)NL * FL  * 2);
  u16* xartB = (u16*)alloc((size_t)NA * FAp * 2);
  u16* tmpL  = (u16*)alloc((size_t)NL * C * 2);
  u16* tmpA  = (u16*)alloc((size_t)NA * C * 2);
  u16* sLb   = (u16*)alloc((size_t)NL * C * 2);
  u16* sAb   = (u16*)alloc((size_t)NA * C * 2);
  u16* qL    = (u16*)alloc((size_t)NL * C * 2);
  u16* qA    = (u16*)alloc((size_t)NA * C * 2);
  u16* kC    = (u16*)alloc((size_t)NL * C * 2);
  u16* kR    = (u16*)alloc((size_t)NL * C * 2);
  u16* kD    = (u16*)alloc((size_t)NA * C * 2);
  u16* vC    = (u16*)alloc((size_t)NL * C * 2);
  u16* vR    = (u16*)alloc((size_t)NL * C * 2);
  u16* vD    = (u16*)alloc((size_t)NA * C * 2);
  u16* gL    = (u16*)alloc((size_t)NL * C * 2);
  u16* gA    = (u16*)alloc((size_t)NA * C * 2);

  float* sF_L = (float*)alloc((size_t)NL * C * 4);
  float* sF_A = (float*)alloc((size_t)NA * C * 4);
  float* aoL  = (float*)alloc((size_t)NL * C * 4);
  float* aoA  = (float*)alloc((size_t)NA * C * 4);
  float* accC = (float*)alloc((size_t)NL * C * 4);
  float* accD = (float*)alloc((size_t)NL * C * 4);
  float* accR = (float*)alloc((size_t)NA * C * 4);
  float* denC = (float*)alloc((size_t)NL * 8 * 4);
  float* denD = (float*)alloc((size_t)NL * 8 * 4);
  float* denR = (float*)alloc((size_t)NA * 8 * 4);
  unsigned int* amxC = (unsigned int*)alloc((size_t)NL * 8 * 4);
  unsigned int* amxD = (unsigned int*)alloc((size_t)NL * 8 * 4);
  unsigned int* amxR = (unsigned int*)alloc((size_t)NA * 8 * 4);
  float* lgC = (float*)alloc((size_t)ECC * 8 * 4);
  float* lgR = (float*)alloc((size_t)ECA * 8 * 4);
  float* lgD = (float*)alloc((size_t)EAC * 8 * 4);

  u16* preW1L = (u16*)alloc((size_t)FL  * C * 2);
  u16* preW2L = (u16*)alloc((size_t)C * C * 2);
  u16* preW1A = (u16*)alloc((size_t)FAp * C * 2);
  u16* preW2A = (u16*)alloc((size_t)C * C * 2);
  u16* poW1A  = (u16*)alloc((size_t)C * C * 2);
  u16* poW2A  = (u16*)alloc((size_t)C * C * 2);
  u16* poW1L  = (u16*)alloc((size_t)C * C * 2);
  u16* poW2L  = (u16*)alloc((size_t)C * C * 2);
  u16* qWA[2], *qWL[2], *aWA[2], *aWL[2];
  for (int l = 0; l < 2; ++l) {
    qWA[l] = (u16*)alloc((size_t)C * C * 2);
    qWL[l] = (u16*)alloc((size_t)C * C * 2);
    aWA[l] = (u16*)alloc((size_t)C * C * 2);
    aWL[l] = (u16*)alloc((size_t)C * C * 2);
  }
  u16* WkC = (u16*)alloc((size_t)C * C * 2);
  u16* WkR = (u16*)alloc((size_t)C * C * 2);
  u16* WkD = (u16*)alloc((size_t)C * C * 2);
  u16* WvC = (u16*)alloc((size_t)C * C * 2);
  u16* WvR = (u16*)alloc((size_t)C * C * 2);
  u16* WvD = (u16*)alloc((size_t)C * C * 2);
  float* bkC = (float*)alloc(C * 4); float* bkR = (float*)alloc(C * 4);
  float* bkD = (float*)alloc(C * 4); float* bvC = (float*)alloc(C * 4);
  float* bvR = (float*)alloc(C * 4); float* bvD = (float*)alloc(C * 4);

  // --- launch helpers -------------------------------------------------------
  auto cvt = [&](const float* src, u16* dst, int sr, int sc, int dr, int dc) {
    long n = (long)dr * dc;
    cvt_pad_bf16<<<dim3((unsigned)((n + 255) / 256)), dim3(256), 0, stream>>>(
        src, dst, sr, sc, dr, dc);
  };
  auto gemm = [&](const u16* A, int M, int K, const u16* W, const float* bias,
                  float* oF, u16* oB, int relu) {
    gemm_bf16_wmma<<<dim3((unsigned)((M + 127) / 128)), dim3(256), 0, stream>>>(
        A, M, K, W, bias, oF, oB, relu);
  };
  auto fold = [&](const float* W, const float* b, const float* rel,
                  u16* Wo, float* bo) {
    fold_rel<<<dim3((129 * 128 + 255) / 256), dim3(256), 0, stream>>>(W, b, rel, Wo, bo);
  };
  auto relation = [&](const int* ei, int E, const u16* krel, const u16* q,
                      const float* prel, float* lg, unsigned int* amx,
                      float* den, const u16* vrel, float* acc) {
    unsigned gw = (unsigned)(((long)E * 32 + 255) / 256);
    unsigned ge = (unsigned)(((long)E * 8 + 255) / 256);
    edge_logits_max<<<dim3(gw), dim3(256), 0, stream>>>(ei, E, krel, q, prel, lg, amx);
    edge_exp<<<dim3(ge), dim3(256), 0, stream>>>(ei, E, lg, amx, den);
    edge_scatter<<<dim3(gw), dim3(256), 0, stream>>>(ei, E, lg, den, vrel, acc);
  };
  auto elems = [&](long n) { return dim3((unsigned)((n + 255) / 256)); };

  // --- weight conversion (f32 -> bf16, K padded to mult of 32) -------------
  cvt(x_law, xlawB, NL, FL, NL, FL);
  cvt(x_art, xartB, NA, FA, NA, FAp);
  cvt(P(PRE + 4), preW1L, FL, C, FL, C);   cvt(P(PRE + 6), preW2L, C, C, C, C);
  cvt(P(PRE + 0), preW1A, FA, C, FAp, C);  cvt(P(PRE + 2), preW2A, C, C, C, C);
  cvt(P(POST + 0), poW1A, C, C, C, C);     cvt(P(POST + 2), poW2A, C, C, C, C);
  cvt(P(POST + 4), poW1L, C, C, C, C);     cvt(P(POST + 6), poW2L, C, C, C, C);
  for (int l = 0; l < 2; ++l) {
    int b = 27 * l;
    cvt(P(b + Q_ART_W), qWA[l], C, C, C, C);
    cvt(P(b + Q_LAW_W), qWL[l], C, C, C, C);
    cvt(P(b + A_ART_W), aWA[l], C, C, C, C);
    cvt(P(b + A_LAW_W), aWL[l], C, C, C, C);
  }

  // --- pre-MLP: state = relu(relu(x@W1+b1)@W2+b2) --------------------------
  gemm(xlawB, NL, FL,  preW1L, P(PRE + 5), nullptr, tmpL, 1);
  gemm(tmpL,  NL, C,   preW2L, P(PRE + 7), sF_L, sLb, 1);
  gemm(xartB, NA, FAp, preW1A, P(PRE + 1), nullptr, tmpA, 1);
  gemm(tmpA,  NA, C,   preW2A, P(PRE + 3), sF_A, sAb, 1);

  for (int l = 0; l < 2; ++l) {
    int b = 27 * l;

    // fold relation matrices into K/V weights (per-node projection)
    fold(P(b + K_LAW_W), P(b + K_LAW_B), P(b + R_CIT_A), WkC, bkC);
    fold(P(b + K_LAW_W), P(b + K_LAW_B), P(b + R_REF_A), WkR, bkR);
    fold(P(b + K_ART_W), P(b + K_ART_B), P(b + R_RED_A), WkD, bkD);
    fold(P(b + V_LAW_W), P(b + V_LAW_B), P(b + R_CIT_M), WvC, bvC);
    fold(P(b + V_LAW_W), P(b + V_LAW_B), P(b + R_REF_M), WvR, bvR);
    fold(P(b + V_ART_W), P(b + V_ART_B), P(b + R_RED_M), WvD, bvD);

    // node projections (all WMMA)
    gemm(sLb, NL, C, qWL[l], P(b + Q_LAW_B), nullptr, qL, 0);
    gemm(sAb, NA, C, qWA[l], P(b + Q_ART_B), nullptr, qA, 0);
    gemm(sLb, NL, C, WkC, bkC, nullptr, kC, 0);
    gemm(sLb, NL, C, WkR, bkR, nullptr, kR, 0);
    gemm(sAb, NA, C, WkD, bkD, nullptr, kD, 0);
    gemm(sLb, NL, C, WvC, bvC, nullptr, vC, 0);
    gemm(sLb, NL, C, WvR, bvR, nullptr, vR, 0);
    gemm(sAb, NA, C, WvD, bvD, nullptr, vD, 0);

    // zero per-relation softmax/scatter state
    (void)hipMemsetAsync(amxC, 0, (size_t)NL * 8 * 4, stream);
    (void)hipMemsetAsync(amxD, 0, (size_t)NL * 8 * 4, stream);
    (void)hipMemsetAsync(amxR, 0, (size_t)NA * 8 * 4, stream);
    (void)hipMemsetAsync(denC, 0, (size_t)NL * 8 * 4, stream);
    (void)hipMemsetAsync(denD, 0, (size_t)NL * 8 * 4, stream);
    (void)hipMemsetAsync(denR, 0, (size_t)NA * 8 * 4, stream);
    (void)hipMemsetAsync(accC, 0, (size_t)NL * C * 4, stream);
    (void)hipMemsetAsync(accD, 0, (size_t)NL * C * 4, stream);
    (void)hipMemsetAsync(accR, 0, (size_t)NA * C * 4, stream);

    // edge phase per relation (segment softmax + weighted scatter)
    relation(ei_cites,    ECC, kC, qL, P(b + R_CIT_P), lgC, amxC, denC, vC, accC);
    relation(ei_referred, EAC, kD, qL, P(b + R_RED_P), lgD, amxD, denD, vD, accD);
    relation(ei_refers,   ECA, kR, qA, P(b + R_REF_P), lgR, amxR, denR, vR, accR);

    // mean over relations + GELU, then a-linear, then gated skip
    combine_gelu<<<elems((long)NL * C), dim3(256), 0, stream>>>(accC, accD, 0.5f, gL, (long)NL * C);
    combine_gelu<<<elems((long)NA * C), dim3(256), 0, stream>>>(accR, nullptr, 1.0f, gA, (long)NA * C);
    gemm(gL, NL, C, aWL[l], P(b + A_LAW_B), aoL, nullptr, 0);
    gemm(gA, NA, C, aWA[l], P(b + A_ART_B), aoA, nullptr, 0);
    skip_gate<<<elems((long)NL * C), dim3(256), 0, stream>>>(aoL, P(b + SKIP_LAW), sF_L, sLb, (long)NL * C);
    skip_gate<<<elems((long)NA * C), dim3(256), 0, stream>>>(aoA, P(b + SKIP_ART), sF_A, sAb, (long)NA * C);

    // post-MLP (params shared across layers, per the reference's aliasing bug)
    gemm(sLb,  NL, C, poW1L, P(POST + 5), nullptr, tmpL, 1);
    gemm(tmpL, NL, C, poW2L, P(POST + 7), sF_L, sLb, 1);
    gemm(sAb,  NA, C, poW1A, P(POST + 1), nullptr, tmpA, 1);
    gemm(tmpA, NA, C, poW2A, P(POST + 3), sF_A, sAb, 1);
  }

  gather_docs<<<dim3((2 * 256 * 128 + 255) / 256), dim3(256), 0, stream>>>(
      sF_L, docA, docB, (float*)d_out);
}
